// CoxPHLoss_43044162241121
// MI455X (gfx1250) — matrix operand order repaired and checked by reading, not compiled, not run on hardware
//
#include <hip/hip_runtime.h>

// ---------------------------------------------------------------------------
// CoxPH partial-likelihood loss, N = 16384.
//   denom[i] = sum_j (time[j] >= time[i]) * exp(clip(risk[j], -20, 20))
//   loss = -sum_i (theta_i - log(denom_i + 1e-8)) * event_i / (sum event + 1e-8)
//
// MI455X plan: all data fits in LDS (time f32 = 64KB, exp bf16 = 32KB of the
// 320KB WGP LDS). The N^2 masked mat-vec runs as v_wmma_f32_16x16x32_bf16:
// the matrix pipe does the MACs while the VALU builds the 0/1 bf16 masks
// (16-bit WMMA co-executes with VALU on CDNA5). BF16 is mandatory: exp(20)
// ~ 4.9e8 overflows FP16 but fits BF16's FP32 exponent range.
// ---------------------------------------------------------------------------

#define COX_N 16384

typedef __bf16 v16bf __attribute__((ext_vector_type(16)));
typedef float  v8f   __attribute__((ext_vector_type(8)));

union TileU {
    unsigned u[8];
    v16bf    v;
};

__device__ __forceinline__ unsigned f32_to_bf16_bits(float x) {
    unsigned u = __float_as_uint(x);
    return (u + 0x7FFFu + ((u >> 16) & 1u)) >> 16;   // round-to-nearest-even
}

// ---------------------------------------------------------------------------
// Kernel 1: denom via BF16 WMMA.
// Grid: 128 blocks x 256 threads (8 wave32 per block). Wave w of block b owns
// row-tile (b*8 + w), i.e. rows [tile*16, tile*16+16).
// Dynamic LDS: s_time[16384] (f32) then s_exp[16384] (bf16 bits).
// ---------------------------------------------------------------------------
__global__ void cox_denom_wmma_kernel(const float* __restrict__ risk,
                                      const float* __restrict__ time,
                                      float* __restrict__ denom) {
    extern __shared__ char smem[];
    float*          s_time = (float*)smem;                              // 64 KB
    unsigned short* s_exp  = (unsigned short*)(smem + COX_N * sizeof(float)); // 32 KB

    const int tid  = threadIdx.x;
    const int lane = tid & 31;
    const int wave = tid >> 5;

    // Stage time + exp(clip(theta)) as bf16 into LDS (whole problem on-chip).
    for (int i = tid; i < COX_N; i += 256) {
        float th = fminf(fmaxf(risk[i], -20.0f), 20.0f);
        s_time[i] = time[i];
        s_exp[i]  = (unsigned short)f32_to_bf16_bits(__expf(th));
    }
    __syncthreads();

    const int tile = blockIdx.x * 8 + wave;      // 128*8 = 1024 tiles = N/16
    // Column n = lane & 15 of the B mask tile; each lane keeps its row time.
    const float tr = s_time[tile * 16 + (lane & 15)];

    const int half = (lane & 16) ? 8 : 0;        // A-matrix 16-bit K split
    const int kb   = (lane & 16) ? 16 : 0;       // unused; B splits K by half, like A

    v8f acc = {};                                 // D = 16x16 f32 (8 VGPRs)

    for (int k0 = 0; k0 < COX_N; k0 += 32) {
        // --- A: exp(theta) chunk broadcast over the 16 rows --------------
        // 16-bit A layout: lane half selects K {0..7}/{8..15} then {16..23}/{24..31}.
        TileU a;
        uint4 e0 = *(const uint4*)&s_exp[k0 + half];        // K = half..half+7
        uint4 e1 = *(const uint4*)&s_exp[k0 + 16 + half];   // K = 16+half..
        a.u[0] = e0.x; a.u[1] = e0.y; a.u[2] = e0.z; a.u[3] = e0.w;
        a.u[4] = e1.x; a.u[5] = e1.y; a.u[6] = e1.z; a.u[7] = e1.w;

        // --- B: risk-set mask tile, built on the VALU pipe ---------------
        // 16-bit B layout: same K order as A (half selects {0..7}/{8..15} then {16..23}/{24..31}).
        TileU b;
#pragma unroll
        for (int w = 0; w < 4; ++w) {
            float4 t = *(const float4*)&s_time[k0 + half + 8 * (w >> 1) + 4 * w];
            b.u[2 * w]     = ((t.x >= tr) ? 0x00003F80u : 0u) |
                             ((t.y >= tr) ? 0x3F800000u : 0u);
            b.u[2 * w + 1] = ((t.z >= tr) ? 0x00003F80u : 0u) |
                             ((t.w >= tr) ? 0x3F800000u : 0u);
        }

        // D += A x B : every D row is identical = partial denom per column.
        acc = __builtin_amdgcn_wmma_f32_16x16x32_bf16(
            /*neg_a=*/false, a.v, /*neg_b=*/false, b.v,
            /*c_mod=*/(short)0, acc, /*reuse_a=*/false, /*reuse_b=*/false);
    }

    // D row M=0 lives in acc[0], lanes 0..15 hold columns n = 0..15.
    if (lane < 16) {
        denom[tile * 16 + lane] = acc[0];
    }
}

// ---------------------------------------------------------------------------
// Kernel 2: loss = -sum((theta - log(denom)) * ev) / (sum ev + 1e-8)
// Single block, fixed-order LDS tree reduction (deterministic).
// ---------------------------------------------------------------------------
__global__ void cox_loss_reduce_kernel(const float* __restrict__ risk,
                                       const int* __restrict__ event,
                                       const float* __restrict__ denom,
                                       float* __restrict__ out) {
    __shared__ float sn[256];
    __shared__ float sd[256];
    const int tid = threadIdx.x;

    float num = 0.0f, den = 0.0f;
    for (int i = tid; i < COX_N; i += 256) {
        float th = fminf(fmaxf(risk[i], -20.0f), 20.0f);
        float ev = (float)event[i];
        float d  = fmaxf(denom[i] + 1e-8f, 1e-8f);
        float ld = __logf(d);
        // nan_to_num(nan=0, posinf=0, neginf=-20)
        if (!(ld == ld))                  ld = 0.0f;
        else if (ld ==  __builtin_inff()) ld = 0.0f;
        else if (ld == -__builtin_inff()) ld = -20.0f;
        num += (th - ld) * ev;
        den += ev;
    }
    sn[tid] = num;
    sd[tid] = den;
    __syncthreads();
    for (int s = 128; s > 0; s >>= 1) {
        if (tid < s) {
            sn[tid] += sn[tid + s];
            sd[tid] += sd[tid + s];
        }
        __syncthreads();
    }
    if (tid == 0) {
        out[0] = -sn[0] / (sd[0] + 1e-8f);
    }
}

// ---------------------------------------------------------------------------
extern "C" void kernel_launch(void* const* d_in, const int* in_sizes, int n_in,
                              void* d_out, int out_size, void* d_ws, size_t ws_size,
                              hipStream_t stream) {
    const float* risk  = (const float*)d_in[0];
    const float* time  = (const float*)d_in[1];
    const int*   event = (const int*)d_in[2];

    float* denom = (float*)d_ws;                 // N * 4 = 64 KB scratch

    const size_t smem_bytes = COX_N * sizeof(float)            // time  (64 KB)
                            + COX_N * sizeof(unsigned short);  // exp   (32 KB)

    // 1024 row-tiles of 16 rows -> 128 blocks x 8 waves.
    cox_denom_wmma_kernel<<<128, 256, smem_bytes, stream>>>(risk, time, denom);
    cox_loss_reduce_kernel<<<1, 256, 0, stream>>>(risk, event, denom, (float*)d_out);
}
